// KLDivLossColBERT_15152644621116
// MI455X (gfx1250) — compile-verified, matrix operand order checked by reading, not run on hardware
//
#include <hip/hip_runtime.h>
#include <hip/hip_bf16.h>
#include <math.h>

typedef __attribute__((ext_vector_type(16))) _Float16 v16h;
typedef __attribute__((ext_vector_type(4)))  _Float16 h4;
typedef __attribute__((ext_vector_type(8)))  float    v8f;

#define B_    128
#define W_    8
#define LQ_   32
#define LD_   180
#define D_    128
#define LDP_  192
#define EPSN_ 1e-12f
#define NEGBIG_ -3.0e38f

// LDS layout (bytes):
//   qA   : [32][128]  _Float16  ->  8192
//   dB   : [192][128] _Float16  -> 49152
//   part : [32][8]    float     ->  1024  (per-row, per-wave max)
//   red  : [1024]     float     ->  4096  (reduction scratch)
//   rowS : [32]       float     ->   128  (query row inverse norms)
//   colS : [128]      float     ->   512  (doc feature-column inverse norms)
//   rowM : [32]       float     ->   128  (final row maxima)
#define SMEM_QA    0
#define SMEM_DB    (SMEM_QA + 32 * 128 * 2)
#define SMEM_PART  (SMEM_DB + 192 * 128 * 2)
#define SMEM_RED   (SMEM_PART + 32 * 8 * 4)
#define SMEM_ROWS  (SMEM_RED + 1024 * 4)
#define SMEM_COLS  (SMEM_ROWS + 32 * 4)
#define SMEM_ROWM  (SMEM_COLS + 128 * 4)
#define SMEM_TOTAL (SMEM_ROWM + 32 * 4)

__global__ __launch_bounds__(256) void colbert_maxsim_wmma_kernel(
    const float* __restrict__ qg,     // [B, LQ, D]
    const float* __restrict__ dg,     // [B, W, LD, D]
    const int*   __restrict__ mg,     // [B, W, LD]
    float*       __restrict__ scores) // [B*W]
{
    extern __shared__ char smem[];
    _Float16* qA   = (_Float16*)(smem + SMEM_QA);
    _Float16* dB   = (_Float16*)(smem + SMEM_DB);
    float*    part = (float*)(smem + SMEM_PART);
    float*    red  = (float*)(smem + SMEM_RED);
    float*    rowS = (float*)(smem + SMEM_ROWS);
    float*    colS = (float*)(smem + SMEM_COLS);
    float*    rowM = (float*)(smem + SMEM_ROWM);

    const int tid = threadIdx.x;
    const int blk = blockIdx.x;       // 0 .. B_*W_-1
    const int b   = blk / W_;
    const int w   = blk % W_;

    const float* qb = qg + (size_t)b * LQ_ * D_;
    const float* db = dg + ((size_t)b * W_ + w) * (size_t)LD_ * D_;
    const int*   mb = mg + ((size_t)b * W_ + w) * LD_;

    const int lane32 = tid & 31;
    const int fq     = lane32 * 4;    // this thread's fixed 4-feature column slice
    const int rbase  = tid >> 5;      // row base (step 8)

    // ---- Phase 1: query load (float4, kept in regs) + row sumsq partials ----
    float4 qv[4];
    {
        const float4* q4 = (const float4*)qb;
        #pragma unroll
        for (int i = 0; i < 4; ++i) {
            const int e4  = tid + 256 * i;        // vec4 index; row = e4>>5
            const int row = rbase + 8 * i;
            float4 v = q4[e4];
            qv[i] = v;
            red[row * 32 + lane32] = v.x * v.x + v.y * v.y + v.z * v.z + v.w * v.w;
        }
    }
    __syncthreads();

    // ---- Phase 2: finish query row inverse norms ----
    if (tid < 32) {
        float s = 0.f;
        #pragma unroll
        for (int j = 0; j < 32; ++j) s += red[tid * 32 + j];
        rowS[tid] = 1.f / fmaxf(sqrtf(s), EPSN_);
    }
    __syncthreads();

    // ---- Phase 3: qA fill from regs; single-pass doc load -> f16 LDS +
    //               per-feature (column) sumsq partials in regs ----
    {
        #pragma unroll
        for (int i = 0; i < 4; ++i) {
            const int row = rbase + 8 * i;
            const float s = rowS[row];
            h4 hv;
            hv[0] = (_Float16)(qv[i].x * s);
            hv[1] = (_Float16)(qv[i].y * s);
            hv[2] = (_Float16)(qv[i].z * s);
            hv[3] = (_Float16)(qv[i].w * s);
            *(h4*)(qA + row * D_ + fq) = hv;
        }
    }
    float cp0 = 0.f, cp1 = 0.f, cp2 = 0.f, cp3 = 0.f;
    {
        const float4* d4 = (const float4*)db;
        #pragma unroll
        for (int i = 0; i < 24; ++i) {
            const int e4 = tid + 256 * i;         // vec4 index into [192][128]
            const int l  = e4 >> 5;               // token row
            h4 hv = {(_Float16)0.f, (_Float16)0.f, (_Float16)0.f, (_Float16)0.f};
            if (l < LD_) {
                float4 v = d4[e4];
                const float mk = mb[l] ? 1.f : 0.f;
                v.x *= mk; v.y *= mk; v.z *= mk; v.w *= mk;
                cp0 += v.x * v.x; cp1 += v.y * v.y;
                cp2 += v.z * v.z; cp3 += v.w * v.w;
                hv[0] = (_Float16)v.x; hv[1] = (_Float16)v.y;
                hv[2] = (_Float16)v.z; hv[3] = (_Float16)v.w;
            }
            *(h4*)(dB + e4 * 4) = hv;
        }
        // publish 4 contiguous column partials: red[(tid>>5)*128 + fq .. +3]
        float4 cp = make_float4(cp0, cp1, cp2, cp3);
        *(float4*)&red[rbase * 128 + fq] = cp;
    }
    __syncthreads();

    // ---- Phase 4: doc feature-column inverse norms ----
    if (tid < 128) {
        float s = 0.f;
        #pragma unroll
        for (int g = 0; g < 8; ++g) s += red[g * 128 + tid];
        colS[tid] = 1.f / fmaxf(sqrtf(s), EPSN_);
    }
    __syncthreads();

    // ---- Phase 5: rescale dB in place; init per-row/per-wave max ----
    {
        const float c0 = colS[fq + 0], c1 = colS[fq + 1];
        const float c2 = colS[fq + 2], c3 = colS[fq + 3];
        #pragma unroll
        for (int i = 0; i < 24; ++i) {
            const int e4 = tid + 256 * i;
            h4 hv = *(h4*)(dB + e4 * 4);
            hv[0] = (_Float16)((float)hv[0] * c0);
            hv[1] = (_Float16)((float)hv[1] * c1);
            hv[2] = (_Float16)((float)hv[2] * c2);
            hv[3] = (_Float16)((float)hv[3] * c3);
            *(h4*)(dB + e4 * 4) = hv;
        }
        part[tid] = NEGBIG_;              // 32x8
    }
    __syncthreads();

    // ---- Phase 6: WMMA GEMM (sim = qA 32x128 * dB^T 128x192) + running max --
    const int wave = tid >> 5;
    const int lane = tid & 31;
    const int grp  = lane >> 4;           // K-half selector
    const int l16  = lane & 15;
    const int mt   = wave >> 2;           // constant m-tile per wave
    const _Float16* arow = qA + (mt * 16 + l16) * D_;

    float rmax[8];
    #pragma unroll
    for (int r = 0; r < 8; ++r) rmax[r] = NEGBIG_;

    #pragma unroll
    for (int j = 0; j < 3; ++j) {         // 3 contiguous n-tiles per wave
        const int nt = (wave & 3) * 3 + j;
        const _Float16* brow = dB + (nt * 16 + l16) * D_;
        v8f c = {0.f, 0.f, 0.f, 0.f, 0.f, 0.f, 0.f, 0.f};
        #pragma unroll
        for (int ks = 0; ks < 4; ++ks) {
            v16h a, bf;
            #pragma unroll
            for (int i = 0; i < 16; ++i) {
                const int v  = i >> 1, s = i & 1;
                const int kl = ((v < 4) ? (2 * v + s) : (16 + 2 * (v - 4) + s))
                               + grp * 8 + ks * 32;
                a[i]  = arow[kl];
                bf[i] = brow[kl];
            }
            c = __builtin_amdgcn_wmma_f32_16x16x32_f16(
                    false, a, false, bf, (short)0, c, false, false);
        }
        // C layout: lane -> column nt*16+l16; VGPR r -> row mt*16 + grp*8 + r
        const int ncol = nt * 16 + l16;
        #pragma unroll
        for (int r = 0; r < 8; ++r)
            rmax[r] = fmaxf(rmax[r], (ncol < LD_) ? c[r] : NEGBIG_);
    }
    // one cross-lane max per row half (xor masks <16 stay inside 16-lane groups)
    #pragma unroll
    for (int r = 0; r < 8; ++r) {
        float v = rmax[r];
        #pragma unroll
        for (int off = 1; off < 16; off <<= 1)
            v = fmaxf(v, __shfl_xor(v, off, 32));
        if (l16 == 0)
            part[(mt * 16 + grp * 8 + r) * 8 + wave] = v;
    }
    __syncthreads();

    // ---- Phase 7: finish row max, sum rows, emit score ----
    if (tid < 32) {
        float m = part[tid * 8];
        #pragma unroll
        for (int j = 1; j < 8; ++j) m = fmaxf(m, part[tid * 8 + j]);
        rowM[tid] = m;
    }
    __syncthreads();
    if (tid == 0) {
        float s = 0.f;
        #pragma unroll
        for (int r = 0; r < 32; ++r) s += rowM[r];
        scores[blk] = s;
    }
}

// log_softmax over W=8 + KLDiv(log_target=True, batchmean) -> scalar
__global__ __launch_bounds__(128) void colbert_kl_reduce_kernel(
    const float* __restrict__ scores,   // [B, W]
    const float* __restrict__ labels,   // [B, W] (log-probs)
    float*       __restrict__ out)      // [1]
{
    __shared__ float partsum[128];
    const int bi = threadIdx.x;
    const float* s = scores + bi * W_;
    const float* t = labels + bi * W_;

    float mx = NEGBIG_;
    #pragma unroll
    for (int w2 = 0; w2 < W_; ++w2) mx = fmaxf(mx, s[w2]);
    float se = 0.f;
    #pragma unroll
    for (int w2 = 0; w2 < W_; ++w2) se += __expf(s[w2] - mx);
    const float lse = __logf(se) + mx;

    float p = 0.f;
    #pragma unroll
    for (int w2 = 0; w2 < W_; ++w2) {
        const float logp = s[w2] - lse;
        p += __expf(t[w2]) * (t[w2] - logp);
    }
    partsum[bi] = p;
    __syncthreads();
    #pragma unroll
    for (int off = 64; off > 0; off >>= 1) {
        if (bi < off) partsum[bi] += partsum[bi + off];
        __syncthreads();
    }
    if (bi == 0) out[0] = partsum[0] / (float)B_;
}

extern "C" void kernel_launch(void* const* d_in, const int* in_sizes, int n_in,
                              void* d_out, int out_size, void* d_ws, size_t ws_size,
                              hipStream_t stream) {
    const float* q   = (const float*)d_in[0];   // [128,32,128] f32
    const float* d   = (const float*)d_in[1];   // [128,8,180,128] f32
    const int*   m   = (const int*)d_in[2];     // [128,8,180] i32
    const float* lab = (const float*)d_in[3];   // [128,8] f32
    float* out    = (float*)d_out;
    float* scores = (float*)d_ws;               // [128*8] f32 scratch

    colbert_maxsim_wmma_kernel<<<B_ * W_, 256, SMEM_TOTAL, stream>>>(q, d, m, scores);
    colbert_kl_reduce_kernel<<<1, 128, 0, stream>>>(scores, lab, out);
}
